// GCN_37907381354729
// MI455X (gfx1250) — matrix operand order repaired and checked by reading, not compiled, not run on hardware
//
#include <hip/hip_runtime.h>

#define ND  50000
#define ED  800000
#define GD  1024
#define DD  300
#define DP  320
#define LL  5
#define FT  256
#define EPSB 1e-5f

typedef __bf16 bf16;
typedef __attribute__((ext_vector_type(16))) bf16  v16bf;
typedef __attribute__((ext_vector_type(8)))  bf16  v8bf;
typedef __attribute__((ext_vector_type(8)))  float v8f;

__device__ __forceinline__ bf16 f2bf(float f) {
    unsigned u = __builtin_bit_cast(unsigned, f);
    u += 0x7fffu + ((u >> 16) & 1u);                 // round-to-nearest-even
    unsigned short s = (unsigned short)(u >> 16);
    return __builtin_bit_cast(bf16, s);
}

// dst[n][k] = (k<Ksrc && n<Nsrc) ? src[k][n] : 0   (transpose + pad + bf16)
__global__ void k_prep_bt(const float* __restrict__ src, bf16* __restrict__ dst,
                          int Ksrc, int Nsrc, int Kpad, int Npad) {
    int idx = blockIdx.x * blockDim.x + threadIdx.x;
    if (idx >= Kpad * Npad) return;
    int n = idx / Kpad, k = idx % Kpad;
    float v = (k < Ksrc && n < Nsrc) ? src[k * Nsrc + n] : 0.f;
    dst[idx] = f2bf(v);
}

// h0 = x_emb1[x[:,0]] + x_emb2[x[:,1]]  -> bf16, padded
__global__ void k_embed(const int* __restrict__ x, const float* __restrict__ emb1,
                        const float* __restrict__ emb2, bf16* __restrict__ hb) {
    int idx = blockIdx.x * blockDim.x + threadIdx.x;
    if (idx >= ND * DP) return;
    int n = idx / DP, d = idx % DP;
    float v = 0.f;
    if (d < DD) v = emb1[x[n * 2 + 0] * DD + d] + emb2[x[n * 2 + 1] * DD + d];
    hb[idx] = f2bf(v);
}

// C = A[M,K]bf16 * B[K,N]bf16 (+bias)(+relu); Bt is B transposed [N,K].
// One wave -> 16x64 output tile via 4x v_wmma_f32_16x16x32_bf16 per K-step.
__global__ __launch_bounds__(256)
void k_gemm(const bf16* __restrict__ A, const bf16* __restrict__ Bt,
            float* __restrict__ Cf, bf16* __restrict__ Cb,
            const float* __restrict__ bias,
            int M, int Ncols, int K, int relu) {
    int wave = blockIdx.x * (blockDim.x >> 5) + (threadIdx.x >> 5);
    int lane = threadIdx.x & 31;
    int ntiles = Ncols >> 6;
    int jobs = (M >> 4) * ntiles;
    if (wave >= jobs) return;                  // uniform per-wave: EXEC all-1 for WMMA
    int mt = wave / ntiles, nt = wave % ntiles;
    int m0 = mt << 4, n0 = nt << 6;
    int half = lane >> 4, l16 = lane & 15;

    const bf16* arow = A + (size_t)(m0 + l16) * K + half * 16;
    v8f acc[4] = {};
    for (int k = 0; k < K; k += 32) {
        v16bf a;
        {
            v8bf lo = *(const v8bf*)(arow + k);          // K  = k+16h .. +7
            v8bf hi = *(const v8bf*)(arow + k + 8);      // K  = k+16h+8 .. +15
            #pragma unroll
            for (int i = 0; i < 8; ++i) { a[i] = lo[i]; a[i + 8] = hi[i]; }
        }
        #pragma unroll
        for (int t = 0; t < 4; ++t) {
            // B fragment: lane group 0 holds K=k..k+15, group 1 K=k+16..k+31 (contig)
            v16bf bfr = *(const v16bf*)(Bt + (size_t)(n0 + t * 16 + l16) * K + k + half * 16);
            acc[t] = __builtin_amdgcn_wmma_f32_16x16x32_bf16(
                false, a, false, bfr, (short)0, acc[t], false, false);
        }
    }
    #pragma unroll
    for (int t = 0; t < 4; ++t) {
        int col = n0 + t * 16 + l16;
        float bv = bias ? bias[col] : 0.f;
        #pragma unroll
        for (int r = 0; r < 8; ++r) {
            int row = m0 + r + half * 8;                 // C/D layout: vgpr r -> M=r(+8)
            float v = acc[t][r] + bv;
            if (relu) v = fmaxf(v, 0.f);
            size_t o = (size_t)row * Ncols + col;
            if (Cf) Cf[o] = v;
            if (Cb) Cb[o] = f2bf(v);
        }
    }
}

// agg = hw + b[l] + self_e  (pad cols -> 0)
__global__ void k_init_agg(const float* __restrict__ hw, const float* __restrict__ b,
                           const float* __restrict__ e1, const float* __restrict__ e2,
                           float* __restrict__ agg, int l) {
    int idx = blockIdx.x * blockDim.x + threadIdx.x;
    if (idx >= ND * DP) return;
    int d = idx % DP;
    float v = 0.f;
    if (d < DD) v = hw[idx] + b[l * DD + d] + e1[l * 5 + 4] + e2[l * 3 + 0];
    agg[idx] = v;
}

// one wave per edge: agg[col] += hw[row] + eemb   (L2-resident atomics)
__global__ void k_edge(const int* __restrict__ ei, const int* __restrict__ ea,
                       const float* __restrict__ e1, const float* __restrict__ e2,
                       const float* __restrict__ hw, float* __restrict__ agg, int l) {
    int wave = blockIdx.x * (blockDim.x >> 5) + (threadIdx.x >> 5);
    int lane = threadIdx.x & 31;
    if (wave >= ED) return;
    int r = ei[wave], c = ei[ED + wave];
    float ee = e1[l * 5 + ea[2 * wave]] + e2[l * 3 + ea[2 * wave + 1]];
    const float* s = hw + (size_t)r * DP;
    float* dst = agg + (size_t)c * DP;
    for (int d = lane; d < DD; d += 32) atomicAdd(dst + d, s[d] + ee);
}

__global__ void k_zero(float* __restrict__ p, int n) {
    int i = blockIdx.x * blockDim.x + threadIdx.x;
    if (i < n) p[i] = 0.f;
}

// per-channel sum / sumsq (blockDim=320, coalesced row-major reads)
__global__ void k_bn_stats(const float* __restrict__ agg, float* __restrict__ stats, int rpb) {
    int d = threadIdx.x;
    int r0 = blockIdx.x * rpb;
    int r1 = r0 + rpb; if (r1 > ND) r1 = ND;
    float s = 0.f, s2 = 0.f;
    for (int r = r0; r < r1; ++r) {
        float v = agg[(size_t)r * DP + d];
        s += v; s2 += v * v;
    }
    if (d < DD) { atomicAdd(&stats[d], s); atomicAdd(&stats[DP + d], s2); }
}

// fold BN into scale/shift
__global__ void k_bn_final(const float* __restrict__ stats, const float* __restrict__ gamma,
                           const float* __restrict__ beta, float* __restrict__ ss, int l) {
    int d = threadIdx.x;
    float sc = 0.f, sh = 0.f;
    if (d < DD) {
        float mu  = stats[d] / (float)ND;
        float var = stats[DP + d] / (float)ND - mu * mu;
        float inv = rsqrtf(var + EPSB);
        float g = gamma[l * DD + d];
        sc = inv * g;
        sh = beta[l * DD + d] - mu * inv * g;
    }
    ss[d] = sc; ss[DP + d] = sh;
}

// h = relu?(agg*scale+shift) -> bf16 (and f32 on last layer, for pooling)
__global__ void k_bn_apply(const float* __restrict__ agg, const float* __restrict__ ss,
                           bf16* __restrict__ hb, float* __restrict__ hf, int relu, int wf) {
    int idx = blockIdx.x * blockDim.x + threadIdx.x;
    if (idx >= ND * DP) return;
    int d = idx % DP;
    float v = 0.f;
    if (d < DD) {
        v = agg[idx] * ss[d] + ss[DP + d];
        if (relu) v = fmaxf(v, 0.f);
    }
    hb[idx] = f2bf(v);
    if (wf) hf[idx] = v;
}

__global__ void k_pool_acc(const int* __restrict__ batch, const float* __restrict__ h,
                           float* __restrict__ gsum, float* __restrict__ gcnt) {
    int wave = blockIdx.x * (blockDim.x >> 5) + (threadIdx.x >> 5);
    int lane = threadIdx.x & 31;
    if (wave >= ND) return;
    int g = batch[wave];
    const float* s = h + (size_t)wave * DP;
    float* dp = gsum + (size_t)g * DP;
    for (int d = lane; d < DD; d += 32) atomicAdd(dp + d, s[d]);
    if (lane == 0) atomicAdd(gcnt + g, 1.f);
}

__global__ void k_pool_div(const float* __restrict__ gsum, const float* __restrict__ gcnt,
                           bf16* __restrict__ hgb) {
    int idx = blockIdx.x * blockDim.x + threadIdx.x;
    if (idx >= GD * DP) return;
    int g = idx / DP, d = idx % DP;
    float v = 0.f;
    if (d < DD) v = gsum[idx] / fmaxf(gcnt[g], 1.f);
    hgb[idx] = f2bf(v);
}

extern "C" void kernel_launch(void* const* d_in, const int* in_sizes, int n_in,
                              void* d_out, int out_size, void* d_ws, size_t ws_size,
                              hipStream_t stream) {
    (void)in_sizes; (void)n_in; (void)out_size; (void)ws_size;
    const int*   x     = (const int*)d_in[0];
    const int*   ei    = (const int*)d_in[1];
    const int*   ea    = (const int*)d_in[2];
    const int*   batch = (const int*)d_in[3];
    const float* emb1  = (const float*)d_in[4];
    const float* emb2  = (const float*)d_in[5];
    const float* W     = (const float*)d_in[6];
    const float* b     = (const float*)d_in[7];
    const float* e1    = (const float*)d_in[8];
    const float* e2    = (const float*)d_in[9];
    const float* gamma = (const float*)d_in[10];
    const float* beta  = (const float*)d_in[11];
    const float* fw    = (const float*)d_in[12];
    const float* fb    = (const float*)d_in[13];
    const float* w1    = (const float*)d_in[14];
    const float* b1    = (const float*)d_in[15];
    const float* w2    = (const float*)d_in[16];
    const float* b2    = (const float*)d_in[17];
    float* out = (float*)d_out;

    size_t off = 0;
    auto alloc = [&](size_t bytes) -> void* {
        void* p = (char*)d_ws + off;
        off += (bytes + 255) & ~(size_t)255;
        return p;
    };
    bf16*  hb    = (bf16*) alloc((size_t)ND * DP * 2);
    float* hw    = (float*)alloc((size_t)ND * DP * 4);
    float* agg   = (float*)alloc((size_t)ND * DP * 4);
    bf16*  WtB   = (bf16*) alloc((size_t)LL * DP * DP * 2);
    bf16*  fwT   = (bf16*) alloc((size_t)FT * DP * 2);
    bf16*  w1T   = (bf16*) alloc((size_t)FT * FT * 2);
    bf16*  w2T   = (bf16*) alloc((size_t)128 * FT * 2);
    float* stats = (float*)alloc(2 * DP * 4);
    float* ss    = (float*)alloc(2 * DP * 4);
    float* gsum  = (float*)alloc((size_t)GD * DP * 4);
    float* gcnt  = (float*)alloc((size_t)GD * 4);
    bf16*  hgb   = (bf16*) alloc((size_t)GD * DP * 2);
    bf16*  hfb   = (bf16*) alloc((size_t)GD * FT * 2);
    bf16*  tb    = (bf16*) alloc((size_t)GD * FT * 2);
    float* tf    = (float*)alloc((size_t)GD * FT * 4);

    const int TB = 256;
    auto cdiv = [](long long a, long long b) { return (int)((a + b - 1) / b); };

    // weight prep: transpose + pad + bf16 (runs every call; deterministic)
    for (int l = 0; l < LL; ++l)
        k_prep_bt<<<cdiv(DP * DP, TB), TB, 0, stream>>>(
            W + (size_t)l * DD * DD, WtB + (size_t)l * DP * DP, DD, DD, DP, DP);
    k_prep_bt<<<cdiv(FT * DP, TB), TB, 0, stream>>>(fw, fwT, DD, FT, DP, FT);
    k_prep_bt<<<cdiv(FT * FT, TB), TB, 0, stream>>>(w1, w1T, FT, FT, FT, FT);
    k_prep_bt<<<cdiv(128 * FT, TB), TB, 0, stream>>>(w2, w2T, FT, 128, FT, 128);

    k_embed<<<cdiv((long long)ND * DP, TB), TB, 0, stream>>>(x, emb1, emb2, hb);

    for (int l = 0; l < LL; ++l) {
        int jobs = (ND / 16) * (DP / 64);
        k_gemm<<<cdiv(jobs, 8), TB, 0, stream>>>(
            hb, WtB + (size_t)l * DP * DP, hw, nullptr, nullptr, ND, DP, DP, 0);
        k_init_agg<<<cdiv((long long)ND * DP, TB), TB, 0, stream>>>(hw, b, e1, e2, agg, l);
        k_edge<<<cdiv(ED, 8), TB, 0, stream>>>(ei, ea, e1, e2, hw, agg, l);
        k_zero<<<cdiv(2 * DP, TB), TB, 0, stream>>>(stats, 2 * DP);
        k_bn_stats<<<cdiv(ND, 128), DP, 0, stream>>>(agg, stats, 128);
        k_bn_final<<<1, DP, 0, stream>>>(stats, gamma, beta, ss, l);
        k_bn_apply<<<cdiv((long long)ND * DP, TB), TB, 0, stream>>>(
            agg, ss, hb, hw, (l < LL - 1) ? 1 : 0, (l == LL - 1) ? 1 : 0);
    }

    // global mean pool
    k_zero<<<cdiv(GD * DP, TB), TB, 0, stream>>>(gsum, GD * DP);
    k_zero<<<1, TB, 0, stream>>>(gcnt, GD);
    k_pool_acc<<<cdiv(ND, 8), TB, 0, stream>>>(batch, hw, gsum, gcnt);
    k_pool_div<<<cdiv(GD * DP, TB), TB, 0, stream>>>(gsum, gcnt, hgb);

    // h_feat = hg @ feat_w + feat_b   -> d_out[0 : G*FT]
    k_gemm<<<cdiv((GD / 16) * (FT / 64), 8), TB, 0, stream>>>(
        hgb, fwT, out, hfb, fb, GD, FT, DP, 0);
    // t = relu(h_feat @ w1 + b1)
    k_gemm<<<cdiv((GD / 16) * (FT / 64), 8), TB, 0, stream>>>(
        hfb, w1T, tf, tb, b1, GD, FT, FT, 1);
    // out = t @ w2 + b2               -> d_out[G*FT : ]
    k_gemm<<<cdiv((GD / 16) * (128 / 64), 8), TB, 0, stream>>>(
        tb, w2T, out + (size_t)GD * FT, nullptr, b2, GD, 128, FT, 0);
}
